// MoREMG2Phoneme_19104014532937
// MI455X (gfx1250) — matrix operand order repaired and checked by reading, not compiled, not run on hardware
//
#include <hip/hip_runtime.h>
#include <hip/hip_bf16.h>

#define N_TOK  8192
#define DMODEL 512
#define DFF    2048
#define NHEADS 8
#define DHEAD  64
#define SEQ    2048
#define BATCH  4
#define TD     1536   // 3*DMODEL

typedef __bf16 bf16_t;
typedef bf16_t v16bf __attribute__((ext_vector_type(16)));
typedef float  v8f   __attribute__((ext_vector_type(8)));
typedef float  v16f  __attribute__((ext_vector_type(16)));
typedef unsigned short u16;
typedef u16    v16u  __attribute__((ext_vector_type(16)));
typedef u16    u16x8 __attribute__((ext_vector_type(8)));

// hardware f32->bf16 (RNE) via native cast
__device__ __forceinline__ u16 f2bf(float f) {
  return __builtin_bit_cast(u16, (bf16_t)f);
}
__device__ __forceinline__ float bf2f(u16 h) {
  unsigned u = ((unsigned)h) << 16;
  return __builtin_bit_cast(float, u);
}
// concat two contiguous 16B bf16 loads into one fragment
__device__ __forceinline__ v16bf ldfrag(const u16* p0, const u16* p1) {
  u16x8 lo = *(const u16x8*)p0;
  u16x8 hi = *(const u16x8*)p1;
  v16u u = __builtin_shufflevector(lo, hi, 0,1,2,3,4,5,6,7,8,9,10,11,12,13,14,15);
  return __builtin_bit_cast(v16bf, u);
}

// ---------------- f32 -> bf16 weight conversion ----------------
__global__ void cvt_bf16_kernel(const float* __restrict__ src, u16* __restrict__ dst, int n) {
  int i = blockIdx.x * blockDim.x + threadIdx.x;
  if (i < n) dst[i] = f2bf(src[i]);
}

// ---------------- LayerNorm (one block per token) -> bf16 ----------------
__global__ __launch_bounds__(256) void ln_kernel(const float* __restrict__ x,
                                                 const float* __restrict__ g,
                                                 const float* __restrict__ bta,
                                                 u16* __restrict__ out) {
  __shared__ float s1[256], s2[256];
  int t = blockIdx.x;
  const float* xr = x + (size_t)t * DMODEL;
  float v0 = xr[threadIdx.x];
  float v1 = xr[threadIdx.x + 256];
  s1[threadIdx.x] = v0 + v1;
  s2[threadIdx.x] = v0 * v0 + v1 * v1;
  __syncthreads();
  for (int s = 128; s > 0; s >>= 1) {
    if (threadIdx.x < s) { s1[threadIdx.x] += s1[threadIdx.x + s]; s2[threadIdx.x] += s2[threadIdx.x + s]; }
    __syncthreads();
  }
  float mu  = s1[0] * (1.0f / DMODEL);
  float var = s2[0] * (1.0f / DMODEL) - mu * mu;
  float rs  = rsqrtf(var + 1e-5f);
  u16* orow = out + (size_t)t * DMODEL;
  orow[threadIdx.x]       = f2bf((v0 - mu) * rs * g[threadIdx.x]       + bta[threadIdx.x]);
  orow[threadIdx.x + 256] = f2bf((v1 - mu) * rs * g[threadIdx.x + 256] + bta[threadIdx.x + 256]);
}

// ---------------- WMMA NT GEMM:  out[m,n] = A[m,:]·W[n,:] + bias[n] (+epilogue) ----
// Wave tile 32(M)x64(N): 2 A-frags x 4 B-frags = 8 WMMA / K-step (B reused 2x).
// EPI 0: f32 out = gemm+bias
// EPI 1: f32 out = gemm+bias+res               (attn out-proj residual)
// EPI 2: bf16 out = gelu(gemm+bias)            (FFN1 / router hidden)
// EPI 3: f32 out = keep? gemm+bias+res : cur   (FFN2 + depth select)
// EPI 4: QKV: q (scaled 0.125) and k -> bf16 rows in outp; v -> transposed vT[b*512+d][seq]
template<int EPI>
__global__ __launch_bounds__(256) void gemm_kernel(const u16* __restrict__ A,
                                                   const u16* __restrict__ W,
                                                   const float* __restrict__ bias,
                                                   const float* __restrict__ res,
                                                   const float* __restrict__ cursrc,
                                                   const unsigned char* __restrict__ depths,
                                                   int level, void* __restrict__ outp,
                                                   u16* __restrict__ vT,
                                                   int M, int N, int K) {
  const int wave = threadIdx.x >> 5;
  const int lane = threadIdx.x & 31;
  const int n15  = lane & 15;
  const int hi   = lane >> 4;
  const int kbA  = hi * 8;
  const int kbB  = hi * 8;
  const int m0   = blockIdx.y * 256 + wave * 32;
  const int bn   = blockIdx.x * 64;

  v8f acc0[4] = {};
  v8f acc1[4] = {};
  const u16* arow0 = A + (size_t)(m0 + n15) * K;
  const u16* arow1 = A + (size_t)(m0 + 16 + n15) * K;
  for (int k0 = 0; k0 < K; k0 += 32) {
    v16bf a0 = ldfrag(arow0 + k0 + kbA, arow0 + k0 + 16 + kbA);
    v16bf a1 = ldfrag(arow1 + k0 + kbA, arow1 + k0 + 16 + kbA);
#pragma unroll
    for (int t = 0; t < 4; ++t) {
      const u16* wrow = W + (size_t)(bn + t * 16 + n15) * K + k0 + kbB;
      v16bf b = ldfrag(wrow, wrow + 16);
      acc0[t] = __builtin_amdgcn_wmma_f32_16x16x32_bf16(
          false, a0, false, b, (short)0, acc0[t], false, false);
      acc1[t] = __builtin_amdgcn_wmma_f32_16x16x32_bf16(
          false, a1, false, b, (short)0, acc1[t], false, false);
    }
  }
#pragma unroll
  for (int mi = 0; mi < 2; ++mi) {
    const int rb = m0 + mi * 16 + hi * 8;
#pragma unroll
    for (int t = 0; t < 4; ++t) {
      int col = bn + t * 16 + n15;
      float bv = bias[col];
#pragma unroll
      for (int j = 0; j < 8; ++j) {
        int r = rb + j;
        float val = (mi ? acc1[t][j] : acc0[t][j]) + bv;
        if (EPI == 0) {
          ((float*)outp)[(size_t)r * N + col] = val;
        } else if (EPI == 1) {
          ((float*)outp)[(size_t)r * N + col] = val + res[(size_t)r * N + col];
        } else if (EPI == 2) {
          float gl = 0.5f * val * (1.0f + erff(val * 0.70710678118654752f));
          ((u16*)outp)[(size_t)r * N + col] = f2bf(gl);
        } else if (EPI == 3) {
          float v2 = val + res[(size_t)r * N + col];
          float cv = cursrc[(size_t)r * N + col];
          ((float*)outp)[(size_t)r * N + col] = (depths[r] >= (unsigned char)level) ? v2 : cv;
        } else {
          if (col < 2 * DMODEL) {
            float sc = (col < DMODEL) ? 0.125f : 1.0f;  // fold softmax 1/sqrt(dh) into q
            ((u16*)outp)[(size_t)r * N + col] = f2bf(val * sc);
          } else {
            int d = col - 2 * DMODEL;                    // 0..511 (h*64+dd)
            int bb = r / SEQ, l = r - bb * SEQ;
            vT[((size_t)bb * DMODEL + d) * SEQ + l] = f2bf(val);
          }
        }
      }
    }
  }
}

// ---------------- fused in-register flash attention --------------------------
// qkv: [B*L, 1536] bf16 rows (q pre-scaled | k | unused). vT: [B*512 + d][seq] bf16.
// One wave per 16 q-rows per (b,h).
__global__ __launch_bounds__(256) void attn_kernel(const u16* __restrict__ qkv,
                                                   const u16* __restrict__ vT,
                                                   u16* __restrict__ obf) {
  const int wave = threadIdx.x >> 5;
  const int lane = threadIdx.x & 31;
  const int b = blockIdx.z, h = blockIdx.y;
  const int r0 = (blockIdx.x * 8 + wave) * 16;
  const int n    = lane & 15;
  const int hi   = lane >> 4;
  const int kb8  = hi * 8;
  const int kb16 = hi * 16;
  const size_t tokbase = (size_t)b * SEQ;

  // Q as B-fragments (already scaled at QKV-GEMM epilogue)
  const u16* qr = qkv + (tokbase + r0 + n) * TD + h * DHEAD;
  v16bf bq0 = ldfrag(qr + kb8,      qr + 16 + kb8);
  v16bf bq1 = ldfrag(qr + 32 + kb8, qr + 48 + kb8);

  // V^T row pointers for this lane's d-columns (t = 0..3)
  const u16* vt0 = vT + ((size_t)b * DMODEL + h * DHEAD +  0 + n) * SEQ + kb8;
  const u16* vt1 = vT + ((size_t)b * DMODEL + h * DHEAD + 16 + n) * SEQ + kb8;
  const u16* vt2 = vT + ((size_t)b * DMODEL + h * DHEAD + 32 + n) * SEQ + kb8;
  const u16* vt3 = vT + ((size_t)b * DMODEL + h * DHEAD + 48 + n) * SEQ + kb8;

  v8f acc0 = {}, acc1 = {}, acc2 = {}, acc3 = {};
  float m_run = -3.4e38f, l_run = 0.f;

  for (int m0 = 0; m0 < SEQ; m0 += 32) {
    // S^T = K_tile @ Q^T for two 16-key blocks (lane = q-row column, regs = key rows)
    const u16* kr0 = qkv + (tokbase + m0 + n) * TD + DMODEL + h * DHEAD;
    const u16* kr1 = qkv + (tokbase + m0 + 16 + n) * TD + DMODEL + h * DHEAD;
    v16bf ka00 = ldfrag(kr0 + kb8,      kr0 + 16 + kb8);
    v16bf ka01 = ldfrag(kr0 + 32 + kb8, kr0 + 48 + kb8);
    v16bf ka10 = ldfrag(kr1 + kb8,      kr1 + 16 + kb8);
    v16bf ka11 = ldfrag(kr1 + 32 + kb8, kr1 + 48 + kb8);
    v8f s0 = {}, s1 = {};
    s0 = __builtin_amdgcn_wmma_f32_16x16x32_bf16(false, ka00, false, bq0, (short)0, s0, false, false);
    s0 = __builtin_amdgcn_wmma_f32_16x16x32_bf16(false, ka01, false, bq1, (short)0, s0, false, false);
    s1 = __builtin_amdgcn_wmma_f32_16x16x32_bf16(false, ka10, false, bq0, (short)0, s1, false, false);
    s1 = __builtin_amdgcn_wmma_f32_16x16x32_bf16(false, ka11, false, bq1, (short)0, s1, false, false);

    // streaming softmax stats (row stats split across lane pair l, l^16)
    float mloc = -3.4e38f;
#pragma unroll
    for (int j = 0; j < 8; ++j) mloc = fmaxf(mloc, fmaxf(s0[j], s1[j]));
    mloc = fmaxf(mloc, __shfl_xor(mloc, 16, 32));
    float mnew = fmaxf(m_run, mloc);
    v16f pf;
#pragma unroll
    for (int j = 0; j < 8; ++j) {
      pf[j]     = __expf(s0[j] - mnew);
      pf[8 + j] = __expf(s1[j] - mnew);
    }
    float sloc = 0.f;
#pragma unroll
    for (int j = 0; j < 16; ++j) sloc += pf[j];
    sloc += __shfl_xor(sloc, 16, 32);
    float alpha = __expf(m_run - mnew);
    l_run = l_run * alpha + sloc;
    m_run = mnew;
    // C(S^T) layout == A(P) layout for K=32 over the two key blocks: pack directly
    v16bf pa = __builtin_convertvector(pf, v16bf);
    // rescale O accumulators (per q-row alpha redistributed by lane broadcast)
#pragma unroll
    for (int j = 0; j < 8; ++j) {
      float aj = __shfl(alpha, kb8 + j, 32);
      acc0[j] *= aj; acc1[j] *= aj; acc2[j] *= aj; acc3[j] *= aj;
    }
    // O += P @ V : V fragments are contiguous rows of vT (keys along memory)
    acc0 = __builtin_amdgcn_wmma_f32_16x16x32_bf16(false, pa, false, ldfrag(vt0 + m0, vt0 + m0 + 16), (short)0, acc0, false, false);
    acc1 = __builtin_amdgcn_wmma_f32_16x16x32_bf16(false, pa, false, ldfrag(vt1 + m0, vt1 + m0 + 16), (short)0, acc1, false, false);
    acc2 = __builtin_amdgcn_wmma_f32_16x16x32_bf16(false, pa, false, ldfrag(vt2 + m0, vt2 + m0 + 16), (short)0, acc2, false, false);
    acc3 = __builtin_amdgcn_wmma_f32_16x16x32_bf16(false, pa, false, ldfrag(vt3 + m0, vt3 + m0 + 16), (short)0, acc3, false, false);
  }
  // normalize and store bf16 attn output
#pragma unroll
  for (int j = 0; j < 8; ++j) {
    float inv = 1.0f / __shfl(l_run, kb8 + j, 32);
    size_t row = tokbase + r0 + kb8 + j;
    u16* orow = obf + row * DMODEL + h * DHEAD + n;
    orow[0]  = f2bf(acc0[j] * inv);
    orow[16] = f2bf(acc1[j] * inv);
    orow[32] = f2bf(acc2[j] * inv);
    orow[48] = f2bf(acc3[j] * inv);
  }
}

// ---------------- router logits (N_out=4) + argmax ----------------
__global__ void router_kernel(const u16* __restrict__ rh, const float* __restrict__ Rw2,
                              const float* __restrict__ Rb2, float* __restrict__ out_depth,
                              float* __restrict__ out_logits, unsigned char* __restrict__ depths) {
  int t = blockIdx.x * blockDim.x + threadIdx.x;
  if (t >= N_TOK) return;
  const u16* hr = rh + (size_t)t * DMODEL;
  float best = -3.4e38f; int bi = 0;
  float lg[4];
#pragma unroll
  for (int o = 0; o < 4; ++o) {
    float acc = Rb2[o];
    const float* w = Rw2 + o * DMODEL;
    for (int k = 0; k < DMODEL; ++k) acc += bf2f(hr[k]) * w[k];
    lg[o] = acc;
    if (acc > best) { best = acc; bi = o; }   // strict > keeps first max (jnp.argmax)
  }
#pragma unroll
  for (int o = 0; o < 4; ++o) out_logits[(size_t)t * 4 + o] = lg[o];
  out_depth[t] = (float)(bi + 1);
  depths[t] = (unsigned char)(bi + 1);
}

// ------------------------------------------------------------------
extern "C" void kernel_launch(void* const* d_in, const int* in_sizes, int n_in,
                              void* d_out, int out_size, void* d_ws, size_t ws_size,
                              hipStream_t stream) {
  (void)in_sizes; (void)n_in; (void)out_size; (void)ws_size;
  const float* x     = (const float*)d_in[0];
  const float* ln1_g = (const float*)d_in[1];
  const float* ln1_b = (const float*)d_in[2];
  const float* Wqkv  = (const float*)d_in[3];
  const float* bqkv  = (const float*)d_in[4];
  const float* Wo    = (const float*)d_in[5];
  const float* bo    = (const float*)d_in[6];
  const float* ln2_g = (const float*)d_in[7];
  const float* ln2_b = (const float*)d_in[8];
  const float* W1    = (const float*)d_in[9];
  const float* b1    = (const float*)d_in[10];
  const float* W2    = (const float*)d_in[11];
  const float* b2    = (const float*)d_in[12];
  const float* rn_g  = (const float*)d_in[13];
  const float* rn_b  = (const float*)d_in[14];
  const float* Rw1   = (const float*)d_in[15];
  const float* Rb1   = (const float*)d_in[16];
  const float* Rw2   = (const float*)d_in[17];
  const float* Rb2   = (const float*)d_in[18];

  char* ws = (char*)d_ws;
  size_t off = 0;
  auto alloc = [&](size_t bytes) {
    off = (off + 255) & ~(size_t)255;
    char* p = ws + off;
    off += bytes;
    return p;
  };
  u16* wqkv_bf = (u16*)alloc((size_t)TD * DMODEL * 2);
  u16* wo_bf   = (u16*)alloc((size_t)DMODEL * DMODEL * 2);
  u16* w1_bf   = (u16*)alloc((size_t)DFF * DMODEL * 2);
  u16* w2_bf   = (u16*)alloc((size_t)DMODEL * DFF * 2);
  u16* rw1_bf  = (u16*)alloc((size_t)DMODEL * DMODEL * 2);
  u16* h_bf    = (u16*)alloc((size_t)N_TOK * DMODEL * 2);
  u16* o_bf    = (u16*)alloc((size_t)N_TOK * DMODEL * 2);   // router hidden, then attn out
  u16* qkv_bf  = (u16*)alloc((size_t)N_TOK * DFF * 2);      // max(qkv 1536, ff 2048) bf16
  u16* ff_bf   = qkv_bf;                                    // aliased after attention
  u16* vT      = (u16*)alloc((size_t)N_TOK * DMODEL * 2);   // V transposed: [b*512+d][seq]
  float* xa    = (float*)alloc((size_t)N_TOK * DMODEL * 4);
  unsigned char* depths = (unsigned char*)alloc(N_TOK);

  float* out_cur    = (float*)d_out;                 // final x lives in d_out
  float* out_depth  = out_cur + (size_t)N_TOK * DMODEL;
  float* out_logits = out_depth + N_TOK;

  // weights -> bf16
  cvt_bf16_kernel<<<(TD * DMODEL + 255) / 256, 256, 0, stream>>>(Wqkv, wqkv_bf, TD * DMODEL);
  cvt_bf16_kernel<<<(DMODEL * DMODEL + 255) / 256, 256, 0, stream>>>(Wo, wo_bf, DMODEL * DMODEL);
  cvt_bf16_kernel<<<(DFF * DMODEL + 255) / 256, 256, 0, stream>>>(W1, w1_bf, DFF * DMODEL);
  cvt_bf16_kernel<<<(DMODEL * DFF + 255) / 256, 256, 0, stream>>>(W2, w2_bf, DMODEL * DFF);
  cvt_bf16_kernel<<<(DMODEL * DMODEL + 255) / 256, 256, 0, stream>>>(Rw1, rw1_bf, DMODEL * DMODEL);

  // router: LN -> GEMM+GELU -> logits/argmax
  ln_kernel<<<N_TOK, 256, 0, stream>>>(x, rn_g, rn_b, h_bf);
  gemm_kernel<2><<<dim3(DMODEL / 64, N_TOK / 256), 256, 0, stream>>>(
      h_bf, rw1_bf, Rb1, nullptr, nullptr, nullptr, 0, o_bf, nullptr, N_TOK, DMODEL, DMODEL);
  router_kernel<<<N_TOK / 256, 256, 0, stream>>>(o_bf, Rw2, Rb2, out_depth, out_logits, depths);

  // cur = x
  hipMemcpyAsync(out_cur, x, (size_t)N_TOK * DMODEL * 4, hipMemcpyDeviceToDevice, stream);

  for (int level = 1; level <= 4; ++level) {
    ln_kernel<<<N_TOK, 256, 0, stream>>>(out_cur, ln1_g, ln1_b, h_bf);
    gemm_kernel<4><<<dim3(TD / 64, N_TOK / 256), 256, 0, stream>>>(
        h_bf, wqkv_bf, bqkv, nullptr, nullptr, nullptr, 0, qkv_bf, vT, N_TOK, TD, DMODEL);
    attn_kernel<<<dim3(SEQ / 128, NHEADS, BATCH), 256, 0, stream>>>(qkv_bf, vT, o_bf);
    gemm_kernel<1><<<dim3(DMODEL / 64, N_TOK / 256), 256, 0, stream>>>(
        o_bf, wo_bf, bo, out_cur, nullptr, nullptr, 0, xa, nullptr, N_TOK, DMODEL, DMODEL);
    ln_kernel<<<N_TOK, 256, 0, stream>>>(xa, ln2_g, ln2_b, h_bf);
    gemm_kernel<2><<<dim3(DFF / 64, N_TOK / 256), 256, 0, stream>>>(
        h_bf, w1_bf, b1, nullptr, nullptr, nullptr, 0, ff_bf, nullptr, N_TOK, DFF, DMODEL);
    gemm_kernel<3><<<dim3(DMODEL / 64, N_TOK / 256), 256, 0, stream>>>(
        ff_bf, w2_bf, b2, xa, out_cur, depths, level, out_cur, nullptr, N_TOK, DMODEL, DFF);
  }
}
